// GNN_13185549598929
// MI455X (gfx1250) — compile-verified, matrix operand order checked by reading, not run on hardware
//
#include <hip/hip_runtime.h>

#define D 64

typedef __attribute__((ext_vector_type(16))) _Float16 v16h;
typedef __attribute__((ext_vector_type(8)))  float    v8f;

union HFrag { v16h v; _Float16 h[16]; };

__device__ __forceinline__ void load8_cvt(const float* __restrict__ p, _Float16* h) {
    const float4 u = *reinterpret_cast<const float4*>(p);
    const float4 w = *reinterpret_cast<const float4*>(p + 4);
    h[0] = (_Float16)u.x; h[1] = (_Float16)u.y; h[2] = (_Float16)u.z; h[3] = (_Float16)u.w;
    h[4] = (_Float16)w.x; h[5] = (_Float16)w.y; h[6] = (_Float16)w.z; h[7] = (_Float16)w.w;
}

// ---------------------------------------------------------------- degree / norm
__global__ void k_init_deg(float* __restrict__ deg, int n) {
    int i = blockIdx.x * blockDim.x + threadIdx.x;
    if (i < n) deg[i] = 1.0f;                       // self-loop contribution
}

__global__ void k_deg_accum(const long long* __restrict__ dst, float* __restrict__ deg, int e) {
    int i = blockIdx.x * blockDim.x + threadIdx.x;
    if (i < e) atomicAdd(&deg[dst[i]], 1.0f);
}

__global__ void k_rsqrt_inplace(float* __restrict__ deg, int n) {
    int i = blockIdx.x * blockDim.x + threadIdx.x;
    if (i < n) {
        float d = deg[i];
        deg[i] = (d > 0.0f) ? rsqrtf(d) : 0.0f;     // becomes dinv
    }
}

// per-edge symmetric normalization, computed once and reused by both layers
__global__ void k_edge_norm(const long long* __restrict__ src,
                            const long long* __restrict__ dst,
                            const float* __restrict__ dinv,
                            float* __restrict__ norm, int e) {
    int i = blockIdx.x * blockDim.x + threadIdx.x;
    if (i < e) norm[i] = dinv[src[i]] * dinv[dst[i]];
}

// ---------------------------------------------------------------- dense H = X @ W^T (WMMA)
// X: [nrows, 64] f32 row-major, W: [64, 64] f32 row-major (W[out][in]).
// One wave computes one 16-row tile x all 64 output columns (4 col-tiles, K=64 in 2 WMMAs).
__global__ void k_gemm64_wmma(const float* __restrict__ X,
                              const float* __restrict__ W,
                              float* __restrict__ H, int nrows) {
    const int lane = threadIdx.x & 31;
    const int wave = threadIdx.x >> 5;
    const int tile = blockIdx.x * 8 + wave;         // 16-row tile, wave-uniform
    const int rowBase = tile * 16;
    if (rowBase >= nrows) return;                   // whole wave exits -> EXEC all-1s at WMMA

    const int m  = lane & 15;
    const int hi = lane >> 4;                       // 0: K 0..7/16..23, 1: K 8..15/24..31

    // A fragments for K=[0,32) and K=[32,64)
    const float* xr = X + (size_t)(rowBase + m) * D;
    HFrag a0, a1;
    load8_cvt(xr +      hi * 8, a0.h);
    load8_cvt(xr + 16 + hi * 8, a0.h + 8);
    load8_cvt(xr + 32 + hi * 8, a1.h);
    load8_cvt(xr + 48 + hi * 8, a1.h + 8);

    #pragma unroll
    for (int c = 0; c < 4; ++c) {                   // output column tiles
        // B column o = c*16 + (lane&15) is W row o (B = W^T), mirrored A layout
        const float* wr = W + (size_t)(c * 16 + m) * D;
        HFrag b0, b1;
        load8_cvt(wr +      hi * 8, b0.h);
        load8_cvt(wr + 16 + hi * 8, b0.h + 8);
        load8_cvt(wr + 32 + hi * 8, b1.h);
        load8_cvt(wr + 48 + hi * 8, b1.h + 8);

        v8f acc = {};
        acc = __builtin_amdgcn_wmma_f32_16x16x32_f16(false, a0.v, false, b0.v,
                                                     (short)0, acc, false, false);
        acc = __builtin_amdgcn_wmma_f32_16x16x32_f16(false, a1.v, false, b1.v,
                                                     (short)0, acc, false, false);

        // D layout: VGPR j -> row (j + 8*hi), col (lane&15)
        float* out = H + (size_t)(rowBase + hi * 8) * D + c * 16 + m;
        #pragma unroll
        for (int j = 0; j < 8; ++j) out[(size_t)j * D] = acc[j];
    }
}

// ---------------------------------------------------------------- helpers
__global__ void k_zero4(float4* __restrict__ p, long long n4) {
    long long i = (long long)blockIdx.x * blockDim.x + threadIdx.x;
    if (i < n4) p[i] = make_float4(0.f, 0.f, 0.f, 0.f);
}

__global__ void k_zero(float* __restrict__ p, int n) {
    int i = blockIdx.x * blockDim.x + threadIdx.x;
    if (i < n) p[i] = 0.0f;
}

// 16 threads per edge, 4 features each: float4 gather + 4 f32 atomics
// agg[dst][4q..4q+3] += norm[e] * H[src][4q..4q+3]
__global__ void k_edge_aggregate4(const long long* __restrict__ src,
                                  const long long* __restrict__ dst,
                                  const float* __restrict__ norm,
                                  const float* __restrict__ H,
                                  float* __restrict__ agg,
                                  long long total) {   // total = E*16
    long long gid = (long long)blockIdx.x * blockDim.x + threadIdx.x;
    if (gid >= total) return;
    const int e = (int)(gid >> 4);
    const int q = (int)(gid & 15);                  // float4 slot within the row
    const long long s = src[e];
    const long long d = dst[e];
    const float w = norm[e];
    const float4 hv = *reinterpret_cast<const float4*>(H + (size_t)s * D + q * 4);
    float* ap = agg + (size_t)d * D + q * 4;
    atomicAdd(ap + 0, w * hv.x);
    atomicAdd(ap + 1, w * hv.y);
    atomicAdd(ap + 2, w * hv.z);
    atomicAdd(ap + 3, w * hv.w);
}

// fold self-loop message + bias + relu; result written in place into agg (float4 wide)
__global__ void k_self_bias_relu4(const float* __restrict__ H,
                                  const float* __restrict__ dinv,
                                  const float* __restrict__ bias,
                                  float* __restrict__ agg, int n) {
    long long gid = (long long)blockIdx.x * blockDim.x + threadIdx.x;
    if (gid >= (long long)n * (D / 4)) return;
    const int node = (int)(gid >> 4);
    const int q    = (int)(gid & 15);
    const float di = dinv[node];
    const float sw = di * di;
    const size_t off = (size_t)node * D + q * 4;
    const float4 hv = *reinterpret_cast<const float4*>(H + off);
    const float4 bv = *reinterpret_cast<const float4*>(bias + q * 4);
    float4 av = *reinterpret_cast<float4*>(agg + off);
    av.x += sw * hv.x + bv.x;  av.y += sw * hv.y + bv.y;
    av.z += sw * hv.z + bv.z;  av.w += sw * hv.w + bv.w;
    av.x = av.x > 0.f ? av.x : 0.f;  av.y = av.y > 0.f ? av.y : 0.f;
    av.z = av.z > 0.f ? av.z : 0.f;  av.w = av.w > 0.f ? av.w : 0.f;
    *reinterpret_cast<float4*>(agg + off) = av;
}

// partial column sums -> pool[64]; blockDim == 64, each block covers 512 rows
__global__ void k_pool(const float* __restrict__ h, float* __restrict__ pool, int n) {
    const int f = threadIdx.x;
    int r0 = blockIdx.x * 512;
    int r1 = r0 + 512; if (r1 > n) r1 = n;
    float s = 0.0f;
    for (int r = r0; r < r1; ++r) s += h[(size_t)r * D + f];
    atomicAdd(&pool[f], s);
}

__global__ void k_fc(const float* __restrict__ pool, const float* __restrict__ fcW,
                     const float* __restrict__ fcb, float* __restrict__ out, int n) {
    __shared__ float s[64];
    const int f = threadIdx.x;                      // blockDim.x == 64
    s[f] = pool[f] * fcW[f];
    __syncthreads();
    for (int off = 32; off > 0; off >>= 1) {
        if (f < off) s[f] += s[f + off];
        __syncthreads();
    }
    if (f == 0) out[0] = s[0] / (float)n + fcb[0];
}

// ---------------------------------------------------------------- launch
extern "C" void kernel_launch(void* const* d_in, const int* in_sizes, int n_in,
                              void* d_out, int out_size, void* d_ws, size_t ws_size,
                              hipStream_t stream) {
    (void)n_in; (void)out_size; (void)ws_size;

    const float*     x    = (const float*)d_in[0];
    const long long* ei   = (const long long*)d_in[1];   // int64 [2, E]
    const float*     W1   = (const float*)d_in[2];
    const float*     b1   = (const float*)d_in[3];
    const float*     W2   = (const float*)d_in[4];
    const float*     b2   = (const float*)d_in[5];
    const float*     fcW  = (const float*)d_in[6];
    const float*     fcb  = (const float*)d_in[7];
    float*           out  = (float*)d_out;

    const int n = in_sizes[0] / D;
    const int e = in_sizes[1] / 2;
    const long long* src = ei;
    const long long* dst = ei + e;

    float* ws   = (float*)d_ws;
    float* dinv = ws;                                // n floats
    float* tmpH = dinv + n;                          // n*D floats (dense projection)
    float* agg  = tmpH + (size_t)n * D;              // n*D floats (accumulator / layer out)
    float* norm = agg  + (size_t)n * D;              // e floats (per-edge weight)
    float* pool = norm + e;                          // 64 floats

    const long long nd  = (long long)n * D;
    const long long nd4 = nd / 4;
    const long long e16 = (long long)e * 16;
    const int nBlk   = (n + 255) / 256;
    const int eBlk   = (e + 255) / 256;
    const int nd4Blk = (int)((nd4 + 255) / 256);
    const int e16Blk = (int)((e16 + 255) / 256);
    const int nTiles = (n + 15) / 16;
    const int gemmBlk = (nTiles + 7) / 8;            // 8 waves (tiles) per 256-thread block

    // degree (with self-loops) -> dinv -> per-edge norm
    k_init_deg<<<nBlk, 256, 0, stream>>>(dinv, n);
    k_deg_accum<<<eBlk, 256, 0, stream>>>(dst, dinv, e);
    k_rsqrt_inplace<<<nBlk, 256, 0, stream>>>(dinv, n);
    k_edge_norm<<<eBlk, 256, 0, stream>>>(src, dst, dinv, norm, e);

    // ---- layer 1
    k_gemm64_wmma<<<gemmBlk, 256, 0, stream>>>(x, W1, tmpH, n);
    k_zero4<<<nd4Blk, 256, 0, stream>>>((float4*)agg, nd4);
    k_edge_aggregate4<<<e16Blk, 256, 0, stream>>>(src, dst, norm, tmpH, agg, e16);
    k_self_bias_relu4<<<nd4Blk, 256, 0, stream>>>(tmpH, dinv, b1, agg, n);   // h1 in agg

    // ---- layer 2 (input h1 = agg)
    k_gemm64_wmma<<<gemmBlk, 256, 0, stream>>>(agg, W2, tmpH, n);
    k_zero4<<<nd4Blk, 256, 0, stream>>>((float4*)agg, nd4);
    k_edge_aggregate4<<<e16Blk, 256, 0, stream>>>(src, dst, norm, tmpH, agg, e16);
    k_self_bias_relu4<<<nd4Blk, 256, 0, stream>>>(tmpH, dinv, b2, agg, n);   // h2 in agg

    // ---- mean pool + FC
    k_zero<<<1, 64, 0, stream>>>(pool, 64);
    k_pool<<<(n + 511) / 512, 64, 0, stream>>>(agg, pool, n);
    k_fc<<<1, 64, 0, stream>>>(pool, fcW, fcb, out, n);
}
